// Attention_58239756533889
// MI455X (gfx1250) — compile-verified, matrix operand order checked by reading, not run on hardware
//
#include <hip/hip_runtime.h>
#include <hip/hip_bf16.h>
#include <math.h>

// ---------------------------------------------------------------------------
// Swin-style windowed attention, fused for MI455X (gfx1250, wave32, WMMA).
//   qkv = x @ Wqkv ; per-head S = (Q Kt)*scale + relbias ; P = softmax(S)
//   O = P V ; out = O @ Wout + b
// All matmuls run on v_wmma_f32_16x16x32_f16 (f16 in, f32 accumulate).
// S never leaves LDS (flash-style fusion). K/V fragments are loop-invariant
// per wave and hoisted into registers; GEMM B-fragments are preloaded in
// register arrays so WMMAs issue back-to-back instead of stalling on L2.
// ---------------------------------------------------------------------------

typedef _Float16 half_t;
typedef __attribute__((ext_vector_type(16))) _Float16 f16x16;
typedef __attribute__((ext_vector_type(8)))  _Float16 f16x8;
typedef __attribute__((ext_vector_type(8)))  float    f32x8;
typedef __attribute__((ext_vector_type(4)))  float    f32x4;

#define NTOK   256
#define HEADS  8
#define HDIM   32
#define INNER  256
#define BATCH  64
static constexpr float kScale = 0.17677669529663687f;   // 32^-0.5

// ---- WMMA helpers ----------------------------------------------------------

__device__ __forceinline__ f32x8 wmma_f16(f16x16 a, f16x16 b, f32x8 c) {
  // (neg_a, A, neg_b, B, c_mod, C, reuse_a, reuse_b)
  return __builtin_amdgcn_wmma_f32_16x16x32_f16(false, a, false, b,
                                                (short)0, c, false, false);
}

// A-fragment (16x32 f16, row-major, row stride ld). ISA 7.12.2 16-bit layout:
// lane<16 : row=lane,    K = 0..7  (v0..3) and 16..23 (v4..7)
// lane>=16: row=lane-16, K = 8..15 (v0..3) and 24..31 (v4..7)
// Same loader serves B-fragments when fed the transposed operand row-major.
__device__ __forceinline__ f16x16 frag_f16(const half_t* base, int ld) {
  const int lane = threadIdx.x & 31;
  const int r = lane & 15, hf = lane >> 4;
  const half_t* p = base + r * ld + hf * 8;
  f16x8 a = *reinterpret_cast<const f16x8*>(p);        // 16B load
  f16x8 b = *reinterpret_cast<const f16x8*>(p + 16);   // 16B load
  f16x16 f;
#pragma unroll
  for (int e = 0; e < 8; ++e) { f[e] = a[e]; f[e + 8] = b[e]; }
  return f;
}

// Same fragment, converting from fp32 source on the fly (for x).
__device__ __forceinline__ f16x16 frag_f32(const float* base, int ld) {
  const int lane = threadIdx.x & 31;
  const int r = lane & 15, hf = lane >> 4;
  const float* p = base + r * ld + hf * 8;
  f32x4 a0 = *reinterpret_cast<const f32x4*>(p);
  f32x4 a1 = *reinterpret_cast<const f32x4*>(p + 4);
  f32x4 b0 = *reinterpret_cast<const f32x4*>(p + 16);
  f32x4 b1 = *reinterpret_cast<const f32x4*>(p + 20);
  f16x16 f;
#pragma unroll
  for (int e = 0; e < 4; ++e) {
    f[e]      = (half_t)a0[e];  f[e + 4]  = (half_t)a1[e];
    f[e + 8]  = (half_t)b0[e];  f[e + 12] = (half_t)b1[e];
  }
  return f;
}

// ---- Kernel 0: weight transpose->f16 + dense bias gather -------------------

__global__ __launch_bounds__(256)
void prep_kernel(const float* __restrict__ w_qkv, const float* __restrict__ w_out,
                 const float* __restrict__ bias_table, const int* __restrict__ rel_index,
                 half_t* __restrict__ wqkvt, half_t* __restrict__ woutt,
                 float* __restrict__ biasg) {
  const int WQKV = 768 * 256, WOUT = 256 * 256, BIAS = HEADS * NTOK * NTOK;
  const int total = WQKV + WOUT + BIAS;
  for (int i = blockIdx.x * blockDim.x + threadIdx.x; i < total;
       i += gridDim.x * blockDim.x) {
    if (i < WQKV) {                        // wqkvt[j][k] = w_qkv[k][j]
      int j = i >> 8, k = i & 255;
      wqkvt[i] = (half_t)w_qkv[k * 768 + j];
    } else if (i < WQKV + WOUT) {          // woutt[j][k] = w_out[k][j]
      int t2 = i - WQKV;
      int j = t2 >> 8, k = t2 & 255;
      woutt[t2] = (half_t)w_out[k * 256 + j];
    } else {                               // biasg[h][i][j]
      int t2 = i - WQKV - WOUT;
      int h = t2 >> 16, ij = t2 & 65535;
      biasg[t2] = bias_table[rel_index[ij] * HEADS + h];
    }
  }
}

// ---- Kernel 1: QKV GEMM (16384x768x256) + head reshape ---------------------
// block = (batch, 16-token tile), 8 waves; wave w does col tiles w,w+8,...
// A-fragments hoisted across all 6 col tiles; per tile the 8 B-fragments are
// preloaded (one load clause, one wait) so the 8 WMMAs issue back-to-back.

__global__ __launch_bounds__(256)
void qkv_kernel(const float* __restrict__ x, const half_t* __restrict__ wqkvt,
                half_t* __restrict__ Qf, half_t* __restrict__ Kf,
                half_t* __restrict__ Vt) {
  const int b    = blockIdx.x >> 4;
  const int rt   = blockIdx.x & 15;
  const int wave = threadIdx.x >> 5;
  const int lane = threadIdx.x & 31;
  const int col  = lane & 15, hf = lane >> 4;

  const float* xbase = x + ((size_t)b * NTOK + rt * 16) * 256;
  f16x16 afr[8];
#pragma unroll
  for (int kk = 0; kk < 8; ++kk) afr[kk] = frag_f32(xbase + kk * 32, 256);

#pragma unroll
  for (int u = 0; u < 6; ++u) {
    const int jt = wave + u * 8;
    f16x16 bfr[8];
#pragma unroll
    for (int kk = 0; kk < 8; ++kk)
      bfr[kk] = frag_f16(wqkvt + (jt * 16) * 256 + kk * 32, 256);
    f32x8 acc = {};
#pragma unroll
    for (int kk = 0; kk < 8; ++kk) acc = wmma_f16(afr[kk], bfr[kk], acc);

    const int jcol = jt * 16;
    if (jcol < 512) {                       // Q or K: [b][h][token][d]
      half_t* dst = (jcol < 256) ? Qf : Kf;
      const int jc = jcol & 255;
      const int head = jc >> 5, dbase = jc & 31;
      half_t* p = dst + ((size_t)(b * HEADS + head) * NTOK + rt * 16 + hf * 8) * HDIM
                      + dbase + col;
#pragma unroll
      for (int v = 0; v < 8; ++v) p[v * HDIM] = (half_t)acc[v];
    } else {                                // V transposed: [b][h][d][token]
      const int jc = jcol - 512;
      const int head = jc >> 5, dbase = jc & 31;
      f16x8 vals;
#pragma unroll
      for (int v = 0; v < 8; ++v) vals[v] = (half_t)acc[v];
      half_t* p = Vt + ((size_t)(b * HEADS + head) * HDIM + dbase + col) * NTOK
                     + rt * 16 + hf * 8;
      *reinterpret_cast<f16x8*>(p) = vals;  // tokens contiguous -> b128 store
    }
  }
}

// ---- Kernel 2: fused attention per (batch, head) ---------------------------
// block = one (b,h), 256 threads / 8 waves, loops 16 query tiles.
// K/V fragments are invariant per wave across the qt loop -> hoisted into
// registers. Softmax is a parallel two-level reduction; P stays unnormalized
// (exp only) and 1/rowsum is folded into the O store.

__global__ __launch_bounds__(256)
void attn_kernel(const half_t* __restrict__ Qf, const half_t* __restrict__ Kf,
                 const half_t* __restrict__ Vt, const float* __restrict__ biasg,
                 half_t* __restrict__ Of) {
  const int bh = blockIdx.x;                 // 0..511
  const int b = bh >> 3, h = bh & 7;
  const int wave = threadIdx.x >> 5;
  const int lane = threadIdx.x & 31;
  const int col  = lane & 15, hf = lane >> 4;
  const int t    = threadIdx.x;

  __shared__ __align__(16) float  S[16][260];     // scores (padded: no conflicts)
  __shared__ __align__(16) half_t P[16][264];     // exp(S - max), f16
  __shared__ __align__(16) float  Oacc[16][32];
  __shared__ float pmax[16][16], psum[16][16];
  __shared__ float rowmax[16], rowinv[16];

  const half_t* Qh = Qf + (size_t)bh * NTOK * HDIM;
  const half_t* Kh = Kf + (size_t)bh * NTOK * HDIM;
  const half_t* Vh = Vt + (size_t)bh * HDIM * NTOK;
  const float*  Bh = biasg + (size_t)h * NTOK * NTOK;

  // Loop-invariant fragments (per wave): two K tiles, two V slices.
  const f16x16 kfr0 = frag_f16(Kh + (wave * 16) * HDIM, HDIM);
  const f16x16 kfr1 = frag_f16(Kh + ((wave + 8) * 16) * HDIM, HDIM);
  const int ct = wave & 1;                   // which 16-wide d tile of O
  const int ks = (wave >> 1) * 2;            // two of eight k-steps
  const f16x16 vfr0 = frag_f16(Vh + (ct * 16) * NTOK + ks * 32, NTOK);
  const f16x16 vfr1 = frag_f16(Vh + (ct * 16) * NTOK + (ks + 1) * 32, NTOK);

  const int r  = t >> 4;                     // row 0..15 (phases 2/4)
  const int cc = t & 15;                     // column lane 0..15

  for (int qt = 0; qt < 16; ++qt) {
    const int qbase = qt * 16;
    // phase 1: S row-block = Q Kt; 8 waves x 2 j-tiles, one WMMA each (d=32)
    {
      f16x16 qfr = frag_f16(Qh + qbase * HDIM, HDIM);
      f32x8 a0 = {}, a1 = {};
      a0 = wmma_f16(qfr, kfr0, a0);
      a1 = wmma_f16(qfr, kfr1, a1);
#pragma unroll
      for (int v = 0; v < 8; ++v) {
        S[v + 8 * hf][wave * 16 + col]       = a0[v];
        S[v + 8 * hf][(wave + 8) * 16 + col] = a1[v];
      }
    }
    __syncthreads();
    // phase 2: S = S*scale + bias; per-thread partial max (column-interleaved:
    // LDS conflict-free, bias loads coalesced). Also zero the O accumulator.
    {
      const float* bp = Bh + (qbase + r) * NTOK;
      float m = -3.0e38f;
#pragma unroll
      for (int i = 0; i < 16; ++i) {
        const int c = cc + 16 * i;
        const float v = S[r][c] * kScale + bp[c];
        S[r][c] = v;
        m = fmaxf(m, v);
      }
      pmax[r][cc] = m;
      Oacc[r][cc] = 0.f;
      Oacc[r][cc + 16] = 0.f;
    }
    __syncthreads();
    // phase 3: fold 16 partial maxes per row
    if (t < 16) {
      float m = pmax[t][0];
#pragma unroll
      for (int i = 1; i < 16; ++i) m = fmaxf(m, pmax[t][i]);
      rowmax[t] = m;
    }
    __syncthreads();
    // phase 4: P = exp(S - rowmax) (unnormalized, f16); partial sums
    {
      const float m = rowmax[r];
      float s = 0.f;
#pragma unroll
      for (int i = 0; i < 16; ++i) {
        const int c = cc + 16 * i;
        const float e = __expf(S[r][c] - m);
        P[r][c] = (half_t)e;
        s += e;
      }
      psum[r][cc] = s;
    }
    __syncthreads();
    // phase 5: fold partial sums -> 1/rowsum
    if (t < 16) {
      float s = psum[t][0];
#pragma unroll
      for (int i = 1; i < 16; ++i) s += psum[t][i];
      rowinv[t] = 1.0f / s;
    }
    __syncthreads();
    // phase 6: O += P*V. 8 waves = 2 d-tiles x 4 k-slices; LDS f32 atomics
    {
      f16x16 p0 = frag_f16(&P[0][0] + ks * 32, 264);
      f16x16 p1 = frag_f16(&P[0][0] + (ks + 1) * 32, 264);
      f32x8 acc = {};
      acc = wmma_f16(p0, vfr0, acc);
      acc = wmma_f16(p1, vfr1, acc);
#pragma unroll
      for (int v = 0; v < 8; ++v)
        atomicAdd(&Oacc[v + 8 * hf][ct * 16 + col], acc[v]);  // ds_add_f32
    }
    __syncthreads();
    // phase 7: normalize by 1/rowsum and store O tile (b, token, h*32+d) f16
    {
      const int e = t * 2;
      const int rr = e >> 5, d = e & 31;
      const float inv = rowinv[rr];
      half_t* p = Of + ((size_t)b * NTOK + qbase + rr) * INNER + h * HDIM + d;
      p[0] = (half_t)(Oacc[rr][d] * inv);
      p[1] = (half_t)(Oacc[rr][d + 1] * inv);
    }
    __syncthreads();
  }
}

// ---- Kernel 3: output projection (16384x256x256) + bias -> fp32 ------------

__global__ __launch_bounds__(256)
void proj_kernel(const half_t* __restrict__ Of, const half_t* __restrict__ woutt,
                 const float* __restrict__ b_out, float* __restrict__ out) {
  const int mt   = blockIdx.x;               // 0..1023 row tiles
  const int wave = threadIdx.x >> 5;
  const int lane = threadIdx.x & 31;
  const int col  = lane & 15, hf = lane >> 4;
  const half_t* abase = Of + (size_t)mt * 16 * 256;

  f16x16 afr[8];
#pragma unroll
  for (int kk = 0; kk < 8; ++kk) afr[kk] = frag_f16(abase + kk * 32, 256);

#pragma unroll
  for (int jj = 0; jj < 2; ++jj) {
    const int jt = wave + jj * 8;
    f16x16 bfr[8];
#pragma unroll
    for (int kk = 0; kk < 8; ++kk)
      bfr[kk] = frag_f16(woutt + (jt * 16) * 256 + kk * 32, 256);
    f32x8 acc = {};
#pragma unroll
    for (int kk = 0; kk < 8; ++kk) acc = wmma_f16(afr[kk], bfr[kk], acc);

    const int c = jt * 16 + col;
    const float bias = b_out[c];
    float* p = out + ((size_t)mt * 16 + hf * 8) * 256 + c;
#pragma unroll
    for (int v = 0; v < 8; ++v) p[(size_t)v * 256] = acc[v] + bias;
  }
}

// ---- launch ----------------------------------------------------------------

extern "C" void kernel_launch(void* const* d_in, const int* in_sizes, int n_in,
                              void* d_out, int out_size, void* d_ws, size_t ws_size,
                              hipStream_t stream) {
  (void)in_sizes; (void)n_in; (void)out_size; (void)ws_size;
  const float* x         = (const float*)d_in[0];
  const float* w_qkv     = (const float*)d_in[1];
  const float* bias_tab  = (const float*)d_in[2];
  const float* w_out     = (const float*)d_in[3];
  const float* b_out     = (const float*)d_in[4];
  const int*   rel_index = (const int*)d_in[5];

  char* ws = (char*)d_ws;
  size_t off = 0;
  auto take = [&](size_t bytes) -> char* {
    char* p = ws + off;
    off = (off + bytes + 255) & ~(size_t)255;
    return p;
  };
  half_t* wqkvt = (half_t*)take((size_t)768 * 256 * sizeof(half_t));     // 384 KB
  half_t* woutt = (half_t*)take((size_t)256 * 256 * sizeof(half_t));     // 128 KB
  half_t* Qf    = (half_t*)take((size_t)BATCH * HEADS * NTOK * HDIM * 2);// 8 MB
  half_t* Kf    = (half_t*)take((size_t)BATCH * HEADS * NTOK * HDIM * 2);// 8 MB
  half_t* Vt    = (half_t*)take((size_t)BATCH * HEADS * NTOK * HDIM * 2);// 8 MB
  half_t* Of    = (half_t*)take((size_t)BATCH * NTOK * INNER * 2);       // 8 MB
  float*  biasg = (float*)take((size_t)HEADS * NTOK * NTOK * 4);         // 2 MB

  prep_kernel<<<1536, 256, 0, stream>>>(w_qkv, w_out, bias_tab, rel_index,
                                        wqkvt, woutt, biasg);
  qkv_kernel<<<BATCH * 16, 256, 0, stream>>>(x, wqkvt, Qf, Kf, Vt);
  attn_kernel<<<BATCH * HEADS, 256, 0, stream>>>(Qf, Kf, Vt, biasg, Of);
  proj_kernel<<<(BATCH * NTOK) / 16, 256, 0, stream>>>(Of, woutt, b_out,
                                                       (float*)d_out);
}
